// ColabCHRONOSNet_25005299597829
// MI455X (gfx1250) — matrix operand order repaired and checked by reading, not compiled, not run on hardware
//
#include <hip/hip_runtime.h>
#include <hip/hip_bf16.h>

typedef __attribute__((ext_vector_type(16))) _Float16 v16h;
typedef __attribute__((ext_vector_type(8)))  _Float16 v8h;
typedef __attribute__((ext_vector_type(8)))  float    v8f;
typedef __attribute__((ext_vector_type(4)))  float    v4f;

#define F_IN  235
#define HID   256
#define HEADS 8

// ---------------- WMMA GEMM:  C[M,N] = act(A[M,K] @ B[K,N] + bias) ----------------
// f32 inputs, f16 multiply via v_wmma_f32_16x16x32_f16, f32 accumulate.
// 256 threads = 8 waves, all stacked along M: BM=256, BN=64; each wave owns a
// 32x64 tile = 2x4 WMMA fragments (8 wmma / 12 ds_load_b128 per K-step).
// LDS tiles are staged directly in the WMMA per-lane fragment layout.
#define BM 256
#define BN 64
#define BK 32
#define TSTR 40   // LDS row stride in halves (80B: 16B aligned, conflict-friendly)

__global__ __launch_bounds__(256) void gemm_wmma_f16(
    const float* __restrict__ A, int lda,
    const float* __restrict__ B, int ldb,
    const float* __restrict__ bias,
    float* __restrict__ C, int ldc,
    int M, int N, int K, int act)
{
  // As[row][slot], slot = grp*16 + j per ISA 16-bit A layout (16x32)
  __shared__ __align__(16) _Float16 As[BM * TSTR];
  // Bs[col][slot], slot = grp*16 + j, same k permutation as As (table order)
  __shared__ __align__(16) _Float16 Bs[BN * TSTR];

  const int tid  = threadIdx.x;
  const int lane = tid & 31;
  const int wid  = tid >> 5;       // 0..7 : wave row block (32 rows each)
  const int m0   = blockIdx.y * BM;
  const int n0   = blockIdx.x * BN;
  const int grp  = lane >> 4;      // lane group 0/1
  const int l16  = lane & 15;

  v8f acc[2][4];
#pragma unroll
  for (int i = 0; i < 2; ++i)
#pragma unroll
    for (int j = 0; j < 4; ++j)
#pragma unroll
      for (int r = 0; r < 8; ++r) acc[i][j][r] = 0.f;

  const bool lda_vec = (lda & 3) == 0;

  for (int kk = 0; kk < K; kk += BK) {
    // ---- stage A tile (256x32): thread t stages the whole row t
    {
      int row  = tid;
      int grow = m0 + row;
      const float* ap = A + (size_t)grow * lda + kk;
      float vals[32];
      if (grow < M && kk + BK <= K) {
        if (lda_vec) {
          const v4f* ap4 = (const v4f*)ap;   // 16B aligned when lda%4==0
#pragma unroll
          for (int q = 0; q < 8; ++q) {
            v4f t = ap4[q];
            vals[q * 4 + 0] = t.x; vals[q * 4 + 1] = t.y;
            vals[q * 4 + 2] = t.z; vals[q * 4 + 3] = t.w;
          }
        } else {
#pragma unroll
          for (int j = 0; j < 32; ++j) vals[j] = ap[j];
        }
      } else {
#pragma unroll
        for (int j = 0; j < 32; ++j) {
          int k = kk + j;
          vals[j] = (grow < M && k < K) ? ap[j] : 0.f;
        }
      }
      // inverse A-layout: k 0..7 ->grp0 j0..7 | 8..15 ->grp1 j0..7
      //                   k 16..23->grp0 j8..15| 24..31->grp1 j8..15
      v8h p0, p1, p2, p3;
#pragma unroll
      for (int j = 0; j < 8; ++j) {
        p0[j] = (_Float16)vals[j];
        p1[j] = (_Float16)vals[8 + j];
        p2[j] = (_Float16)vals[16 + j];
        p3[j] = (_Float16)vals[24 + j];
      }
      _Float16* ab = As + row * TSTR;
      *(v8h*)(ab + 0)  = p0;
      *(v8h*)(ab + 16) = p1;
      *(v8h*)(ab + 8)  = p2;
      *(v8h*)(ab + 24) = p3;
    }
    // ---- stage B tile (32x64) transposed: thread t -> col = t&63, kc = (t>>6)*8
    {
      int col = tid & 63;
      int kc  = (tid >> 6) * 8;
      const float* bp = B + (size_t)(kk + kc) * ldb + n0 + col;
      v8h pk;
      if (kk + kc + 8 <= K) {
#pragma unroll
        for (int i = 0; i < 8; ++i) pk[i] = (_Float16)bp[(size_t)i * ldb];
      } else {
#pragma unroll
        for (int i = 0; i < 8; ++i) {
          int k = kk + kc + i;
          pk[i] = (_Float16)((k < K) ? bp[(size_t)i * ldb] : 0.f);
        }
      }
      *(v8h*)(Bs + col * TSTR + ((kc == 8) ? 16 : (kc == 16) ? 8 : kc)) = pk;
    }
    __syncthreads();

    v16h afrag[2], bfrag[4];
#pragma unroll
    for (int tm = 0; tm < 2; ++tm) {
      int row = wid * 32 + tm * 16 + l16;
      const v8h* pa = (const v8h*)(As + row * TSTR + grp * 16);
      v8h a0 = pa[0], a1 = pa[1];
      afrag[tm] = __builtin_shufflevector(a0, a1, 0, 1, 2, 3, 4, 5, 6, 7,
                                          8, 9, 10, 11, 12, 13, 14, 15);
    }
#pragma unroll
    for (int tn = 0; tn < 4; ++tn) {
      int col = tn * 16 + l16;
      const v8h* pb = (const v8h*)(Bs + col * TSTR + grp * 16);
      v8h b0 = pb[0], b1 = pb[1];
      bfrag[tn] = __builtin_shufflevector(b0, b1, 0, 1, 2, 3, 4, 5, 6, 7,
                                          8, 9, 10, 11, 12, 13, 14, 15);
    }

#pragma unroll
    for (int tm = 0; tm < 2; ++tm)
#pragma unroll
      for (int tn = 0; tn < 4; ++tn)
        acc[tm][tn] = __builtin_amdgcn_wmma_f32_16x16x32_f16(
            false, afrag[tm], false, bfrag[tn], (short)0, acc[tm][tn],
            false, false);
    __syncthreads();
  }

  // C/D layout: vgpr r, lanes0-15 -> M=r, lanes16-31 -> M=r+8; N = lane%16
#pragma unroll
  for (int tm = 0; tm < 2; ++tm) {
#pragma unroll
    for (int tn = 0; tn < 4; ++tn) {
      int col = n0 + tn * 16 + l16;
#pragma unroll
      for (int r = 0; r < 8; ++r) {
        int row = m0 + wid * 32 + tm * 16 + r + grp * 8;
        if (row < M) {
          float v = acc[tm][tn][r];
          if (bias) v += bias[col];
          if (act == 1) v = v > 0.f ? v : 0.f;
          C[(size_t)row * ldc + col] = v;
        }
      }
    }
  }
}

// ---------------- helpers ----------------
__device__ __forceinline__ unsigned fkey(float f) {
  unsigned u = __float_as_uint(f);
  return (u & 0x80000000u) ? ~u : (u | 0x80000000u);
}
__device__ __forceinline__ float funkey(unsigned k) {
  return __uint_as_float((k & 0x80000000u) ? (k & 0x7fffffffu) : ~k);
}

__global__ void fill_f(float* p, float v, int n) {
  int i = blockIdx.x * blockDim.x + threadIdx.x;
  if (i < n) p[i] = v;
}
__global__ void fill_u(unsigned* p, unsigned v, int n) {
  int i = blockIdx.x * blockDim.x + threadIdx.x;
  if (i < n) p[i] = v;
}
__global__ void init_bias_rows(float* out, const float* __restrict__ bias, int total, int C) {
  int i = blockIdx.x * blockDim.x + threadIdx.x;
  if (i < total) out[i] = bias[i % C];
}

// a[n,h] = sum_c hp[n, h*C+c] * att[h*C+c]
__global__ void att_scores(const float* __restrict__ hp, const float* __restrict__ att,
                           float* __restrict__ out, int N, int H, int C) {
  int idx = blockIdx.x * blockDim.x + threadIdx.x;
  if (idx >= N * H) return;
  int n = idx / H, h = idx - n * H;
  const float* hv = hp + (size_t)n * H * C + h * C;
  const float* av = att + h * C;
  float s = 0.f;
  for (int c = 0; c < C; ++c) s += hv[c] * av[c];
  out[idx] = s;
}

__global__ void edge_max(const int* __restrict__ src, const int* __restrict__ dst,
                         int E, int Etot, const float* __restrict__ as_,
                         const float* __restrict__ ad_, unsigned* mkey, int H) {
  int idx = blockIdx.x * blockDim.x + threadIdx.x;
  if (idx >= Etot * H) return;
  int e = idx / H, h = idx - e * H;
  int sN = (e < E) ? src[e] : (e - E);
  int dN = (e < E) ? dst[e] : (e - E);
  float v = as_[sN * H + h] + ad_[dN * H + h];
  v = v > 0.f ? v : 0.2f * v;                 // leaky_relu 0.2
  atomicMax(&mkey[dN * H + h], fkey(v));
}

__global__ void edge_expsum(const int* __restrict__ src, const int* __restrict__ dst,
                            int E, int Etot, const float* __restrict__ as_,
                            const float* __restrict__ ad_, const unsigned* __restrict__ mkey,
                            float* __restrict__ exb, float* ssum, int H) {
  int idx = blockIdx.x * blockDim.x + threadIdx.x;
  if (idx >= Etot * H) return;
  int e = idx / H, h = idx - e * H;
  int sN = (e < E) ? src[e] : (e - E);
  int dN = (e < E) ? dst[e] : (e - E);
  float v = as_[sN * H + h] + ad_[dN * H + h];
  v = v > 0.f ? v : 0.2f * v;
  float ex = expf(v - funkey(mkey[dN * H + h]));
  exb[idx] = ex;
  atomicAdd(&ssum[dN * H + h], ex);
}

__global__ void edge_scatter(const int* __restrict__ src, const int* __restrict__ dst,
                             int E, int Etot, const float* __restrict__ exb,
                             const float* __restrict__ ssum, const float* __restrict__ hp,
                             float* out, int H, int C, float scale) {
  int idx = blockIdx.x * blockDim.x + threadIdx.x;
  if (idx >= Etot * H) return;
  int e = idx / H, h = idx - e * H;
  int sN = (e < E) ? src[e] : (e - E);
  int dN = (e < E) ? dst[e] : (e - E);
  float alpha = exb[idx] / (ssum[dN * H + h] + 1e-16f) * scale;
  const float* hv = hp + (size_t)sN * H * C + h * C;
  float* ov = out + (size_t)dN * H * C + h * C;
  for (int c = 0; c < C; ++c) atomicAdd(&ov[c], alpha * hv[c]);
}

// column mean / rsqrt(var+eps) over rows (population variance, matches jnp var ddof=0)
__global__ __launch_bounds__(256) void bn_stats(const float* __restrict__ x, int ldx,
                                                int Nrows, float* mu, float* rinv, float eps) {
  int c = blockIdx.x;
  __shared__ float s1[256], s2[256];
  float a = 0.f, b = 0.f;
  for (int r = threadIdx.x; r < Nrows; r += 256) {
    float v = x[(size_t)r * ldx + c];
    a += v; b += v * v;
  }
  s1[threadIdx.x] = a; s2[threadIdx.x] = b;
  __syncthreads();
  for (int off = 128; off > 0; off >>= 1) {
    if (threadIdx.x < off) {
      s1[threadIdx.x] += s1[threadIdx.x + off];
      s2[threadIdx.x] += s2[threadIdx.x + off];
    }
    __syncthreads();
  }
  if (threadIdx.x == 0) {
    float m = s1[0] / (float)Nrows;
    float var = s2[0] / (float)Nrows - m * m;
    mu[c] = m;
    rinv[c] = rsqrtf(var + eps);
  }
}

__global__ void bn_apply(const float* __restrict__ x, int ldx, float* y, int ldy,
                         const float* __restrict__ mu, const float* __restrict__ rinv,
                         const float* __restrict__ g, const float* __restrict__ b,
                         int Nrows, int C, int elu) {
  int idx = blockIdx.x * blockDim.x + threadIdx.x;
  if (idx >= Nrows * C) return;
  int r = idx / C, c = idx - r * C;
  float v = (x[(size_t)r * ldx + c] - mu[c]) * rinv[c] * g[c] + b[c];
  if (elu) v = v > 0.f ? v : expf(v) - 1.f;
  y[(size_t)r * ldy + c] = v;
}

// out[n,0:2] = c1[n,:] @ wc2 + bc2
__global__ void final_fc(const float* __restrict__ c1, const float* __restrict__ w,
                         const float* __restrict__ b, float* out, int Nrows) {
  int n = blockIdx.x * blockDim.x + threadIdx.x;
  if (n >= Nrows) return;
  const float* row = c1 + (size_t)n * HID;
  float s0 = b[0], s1 = b[1];
  for (int k = 0; k < HID; ++k) {
    float v = row[k];
    s0 += v * w[k * 2];
    s1 += v * w[k * 2 + 1];
  }
  out[n * 2] = s0;
  out[n * 2 + 1] = s1;
}

// ---------------- host launch ----------------
static inline int ceil_div(int a, int b) { return (a + b - 1) / b; }

extern "C" void kernel_launch(void* const* d_in, const int* in_sizes, int n_in,
                              void* d_out, int out_size, void* d_ws, size_t ws_size,
                              hipStream_t stream) {
  (void)n_in; (void)out_size; (void)ws_size;

  const float* x   = (const float*)d_in[0];
  const int*   ei  = (const int*)d_in[1];
  const float* w_in = (const float*)d_in[2];
  const float* b_in = (const float*)d_in[3];
  const float* wt1 = (const float*)d_in[4];
  const float* bt1 = (const float*)d_in[5];
  const float* wt2 = (const float*)d_in[6];
  const float* bt2 = (const float*)d_in[7];
  const float* gw[3]  = {(const float*)d_in[8],  (const float*)d_in[14], (const float*)d_in[20]};
  const float* gas[3] = {(const float*)d_in[9],  (const float*)d_in[15], (const float*)d_in[21]};
  const float* gad[3] = {(const float*)d_in[10], (const float*)d_in[16], (const float*)d_in[22]};
  const float* gb[3]  = {(const float*)d_in[11], (const float*)d_in[17], (const float*)d_in[23]};
  const float* ng[3]  = {(const float*)d_in[12], (const float*)d_in[18], (const float*)d_in[24]};
  const float* nb[3]  = {(const float*)d_in[13], (const float*)d_in[19], (const float*)d_in[25]};
  const float* wc1 = (const float*)d_in[26];
  const float* bc1 = (const float*)d_in[27];
  const float* wc2 = (const float*)d_in[28];
  const float* bc2 = (const float*)d_in[29];

  const int N = in_sizes[0] / F_IN;
  const int E = in_sizes[1] / 2;
  const int Etot = E + N;
  const int* src = ei;
  const int* dst = ei + E;

  // workspace layout (floats)
  float* ws = (float*)d_ws;
  float* h    = ws; ws += (size_t)N * HID;
  float* t1   = ws; ws += (size_t)N * HID;
  float* hg   = ws; ws += (size_t)N * HID;
  float* hp   = ws; ws += (size_t)N * HID;
  float* outb = ws; ws += (size_t)N * HID;
  float* z    = ws; ws += (size_t)N * 2 * HID;
  float* c1   = ws; ws += (size_t)N * HID;
  float* as_  = ws; ws += (size_t)N * HEADS;
  float* ad_  = ws; ws += (size_t)N * HEADS;
  unsigned* mk = (unsigned*)ws; ws += (size_t)N * HEADS;
  float* ssum = ws; ws += (size_t)N * HEADS;
  float* exb  = ws; ws += (size_t)Etot * HEADS;
  float* mu   = ws; ws += HID;
  float* rinv = ws; ws += HID;

  auto gemm = [&](const float* A, int lda, const float* B, int ldb, const float* bias,
                  float* C, int ldc, int M, int Ncols, int K, int act) {
    dim3 grid(Ncols / BN, ceil_div(M, BM));
    gemm_wmma_f16<<<grid, 256, 0, stream>>>(A, lda, B, ldb, bias, C, ldc, M, Ncols, K, act);
  };

  // h = x @ w_in + b_in
  gemm(x, F_IN, w_in, HID, b_in, h, HID, N, HID, F_IN, 0);
  // temporal MLP: t1 = relu(h@wt1+bt1); h_temp = relu(t1@wt2+bt2) -> z[:,256:]
  gemm(h, HID, wt1, HID, bt1, t1, HID, N, HID, HID, 1);
  gemm(t1, HID, wt2, HID, bt2, z + HID, 2 * HID, N, HID, HID, 1);

  const int neh = Etot * HEADS;
  const int nnh = N * HEADS;
  const float* gin = h;
  for (int l = 0; l < 3; ++l) {
    if (l < 2) {
      const int C = HID / HEADS;  // 32
      gemm(gin, HID, gw[l], HID, nullptr, hp, HID, N, HID, HID, 0);
      att_scores<<<ceil_div(nnh, 256), 256, 0, stream>>>(hp, gas[l], as_, N, HEADS, C);
      att_scores<<<ceil_div(nnh, 256), 256, 0, stream>>>(hp, gad[l], ad_, N, HEADS, C);
      fill_u<<<ceil_div(nnh, 256), 256, 0, stream>>>(mk, 0u, nnh);
      fill_f<<<ceil_div(nnh, 256), 256, 0, stream>>>(ssum, 0.f, nnh);
      init_bias_rows<<<ceil_div(N * HID, 256), 256, 0, stream>>>(outb, gb[l], N * HID, HID);
      edge_max<<<ceil_div(neh, 256), 256, 0, stream>>>(src, dst, E, Etot, as_, ad_, mk, HEADS);
      edge_expsum<<<ceil_div(neh, 256), 256, 0, stream>>>(src, dst, E, Etot, as_, ad_, mk, exb, ssum, HEADS);
      edge_scatter<<<ceil_div(neh, 256), 256, 0, stream>>>(src, dst, E, Etot, exb, ssum, hp, outb, HEADS, C, 1.f);
    } else {
      // layer 2: per-head projection (out=256/head), mean over heads folded into scale
      init_bias_rows<<<ceil_div(N * HID, 256), 256, 0, stream>>>(outb, gb[2], N * HID, HID);
      for (int hh = 0; hh < HEADS; ++hh) {
        gemm(gin, HID, gw[2] + hh * HID, HEADS * HID, nullptr, hp, HID, N, HID, HID, 0);
        att_scores<<<ceil_div(N, 256), 256, 0, stream>>>(hp, gas[2] + hh * HID, as_, N, 1, HID);
        att_scores<<<ceil_div(N, 256), 256, 0, stream>>>(hp, gad[2] + hh * HID, ad_, N, 1, HID);
        fill_u<<<ceil_div(N, 256), 256, 0, stream>>>(mk, 0u, N);
        fill_f<<<ceil_div(N, 256), 256, 0, stream>>>(ssum, 0.f, N);
        edge_max<<<ceil_div(Etot, 256), 256, 0, stream>>>(src, dst, E, Etot, as_, ad_, mk, 1);
        edge_expsum<<<ceil_div(Etot, 256), 256, 0, stream>>>(src, dst, E, Etot, as_, ad_, mk, exb, ssum, 1);
        edge_scatter<<<ceil_div(Etot, 256), 256, 0, stream>>>(src, dst, E, Etot, exb, ssum, hp, outb, 1, HID, 1.f / HEADS);
      }
    }
    bn_stats<<<HID, 256, 0, stream>>>(outb, HID, N, mu, rinv, 1e-5f);
    if (l < 2) {
      bn_apply<<<ceil_div(N * HID, 256), 256, 0, stream>>>(outb, HID, hg, HID, mu, rinv, ng[l], nb[l], N, HID, 1);
      gin = hg;
    } else {
      bn_apply<<<ceil_div(N * HID, 256), 256, 0, stream>>>(outb, HID, z, 2 * HID, mu, rinv, ng[2], nb[2], N, HID, 0);
    }
  }

  // classifier
  gemm(z, 2 * HID, wc1, HID, bc1, c1, HID, N, HID, 2 * HID, 1);
  final_fc<<<ceil_div(N, 256), 256, 0, stream>>>(c1, wc2, bc2, (float*)d_out, N);
}
